// Grace_79637283603151
// MI455X (gfx1250) — matrix operand order repaired and checked by reading, not compiled, not run on hardware
//
#include <hip/hip_runtime.h>
#include <hip/hip_bf16.h>
#include <math.h>

// ---------------------------------------------------------------------------
// Types / helpers
// ---------------------------------------------------------------------------
typedef __bf16 bf16_t;
typedef __attribute__((ext_vector_type(16))) __bf16 v16bf;
typedef __attribute__((ext_vector_type(8)))  float  v8f;

union frag16 { uint4 q[2]; v16bf v; };

__device__ __forceinline__ bf16_t f2bf(float f) {
  union { float f; unsigned u; } v; v.f = f;
  unsigned r = v.u + 0x7FFFu + ((v.u >> 16) & 1u);       // round-nearest-even
  unsigned short h = (unsigned short)(r >> 16);
  union { unsigned short u; bf16_t b; } o; o.u = h; return o.b;
}
__device__ __forceinline__ float bf2f(bf16_t b) {
  union { unsigned short u; bf16_t b; } i; i.b = b;
  union { float f; unsigned u; } o; o.u = ((unsigned)i.u) << 16; return o.f;
}

__device__ __forceinline__ float block_reduce_sum_256(float v) {
  __shared__ float sm[8];
  #pragma unroll
  for (int m = 16; m >= 1; m >>= 1) v += __shfl_xor(v, m, 32);
  int wv = threadIdx.x >> 5, ln = threadIdx.x & 31;
  if (ln == 0) sm[wv] = v;
  __syncthreads();
  if (wv == 0) {
    float t = (ln < 8) ? sm[ln] : 0.0f;
    #pragma unroll
    for (int m = 4; m >= 1; m >>= 1) t += __shfl_xor(t, m, 32);
    if (ln == 0) sm[0] = t;
  }
  __syncthreads();
  float r = sm[0];
  __syncthreads();
  return r;
}

// ---------------------------------------------------------------------------
// Utility kernels
// ---------------------------------------------------------------------------
__global__ void zero_kernel(float* __restrict__ p, int n) {
  int i = blockIdx.x * blockDim.x + threadIdx.x;
  if (i < n) p[i] = 0.0f;
}

__global__ void f32_to_bf16_kernel(const float* __restrict__ src,
                                   bf16_t* __restrict__ dst, int n) {
  int i = blockIdx.x * blockDim.x + threadIdx.x;
  if (i < n) dst[i] = f2bf(src[i]);
}

// dst[C x R] = transpose(src[R x C]) with f32 -> bf16
__global__ void transpose_f32_to_bf16_kernel(const float* __restrict__ src,
                                             bf16_t* __restrict__ dst,
                                             int R, int C) {
  int i = blockIdx.x * blockDim.x + threadIdx.x;
  if (i < R * C) {
    int c = i / R, r = i - c * R;          // dst linear = c*R + r
    dst[i] = f2bf(src[(size_t)r * C + c]);
  }
}

// ---------------------------------------------------------------------------
// K-tile geometry: KB=64 per LDS tile, split into two 32-wide fragment
// groups that are software-pipelined (group1 LDS loads overlap group0 WMMAs).
// Padded LDS row stride 72 bf16 (36 dwords): conflict-free, 16B-aligned slots.
// ---------------------------------------------------------------------------
#define KB  64
#define KP  72
#define GBM 128
#define GBN 64

// ---------------------------------------------------------------------------
// WMMA GEMM with bias (+optional ELU):  C[MxN] = act(A[MxK] @ Bt[NxK]^T + bias)
// A row-major, Bt row-major (row n of Bt = column n of B).
// Block tile 128x64, 8 waves, wave tile 16x64. Double-buffered K loop.
// ---------------------------------------------------------------------------
template <bool ELU, bool OUT_BF16>
__global__ __launch_bounds__(256) void wmma_gemm_bias_kernel(
    const bf16_t* __restrict__ A, const bf16_t* __restrict__ Bt,
    const float* __restrict__ bias, void* __restrict__ Cout,
    int M, int N, int K)
{
  __shared__ __align__(128) bf16_t As[2][GBM][KP];
  __shared__ __align__(128) bf16_t Bs[2][GBN][KP];

  const int tid  = threadIdx.x;
  const int wave = tid >> 5;
  const int lane = tid & 31;
  const int lg   = lane >> 4;      // lane group (0/1)
  const int ln   = lane & 15;
  const int rowBase = blockIdx.x * GBM;
  const int colBase = blockIdx.y * GBN;

  // loaders: A tile 128x64 = 512 x 32B slots (2/thread); B tile 64x64 = 256 (1/thread)
  const int arow = tid >> 2, aq = tid & 3;
  const bf16_t* gA0 = A  + (size_t)(rowBase + arow) * K + aq * 16;
  const bf16_t* gA1 = gA0 + (size_t)64 * K;
  const bf16_t* gB0 = Bt + (size_t)(colBase + arow % 64) * K + aq * 16; // arow<64 only used
  const int brow = tid >> 2, bq = tid & 3;
  const bf16_t* gB  = Bt + (size_t)(colBase + brow) * K + bq * 16;
  (void)gB0;

  v8f acc[4];
  #pragma unroll
  for (int t = 0; t < 4; ++t) acc[t] = (v8f){0,0,0,0,0,0,0,0};

  const int nk = K / KB;
  {
    *((uint4*)(&As[0][arow][aq * 16]))      = *(const uint4*)(gA0);
    *((uint4*)(&As[0][arow + 64][aq * 16])) = *(const uint4*)(gA1);
    *((uint4*)(&Bs[0][brow][bq * 16]))      = *(const uint4*)(gB);
  }
  __syncthreads();

  int cur = 0;
  for (int kk = 0; kk < nk; ++kk) {
    uint4 pa0, pa1, pb;
    const bool more = (kk + 1) < nk;
    if (more) {
      pa0 = *(const uint4*)(gA0 + (kk + 1) * KB);
      pa1 = *(const uint4*)(gA1 + (kk + 1) * KB);
      pb  = *(const uint4*)(gB  + (kk + 1) * KB);
      if (kk + 2 < nk) __builtin_prefetch(gA0 + (kk + 2) * KB, 0, 1);
    }

    // load BOTH fragment groups; group1 regs stay live across group0's WMMAs
    frag16 af[2], bfr[2][4];
    const int m = wave * 16 + ln;
    #pragma unroll
    for (int g = 0; g < 2; ++g) {
      af[g].q[0] = *(const uint4*)(&As[cur][m][g * 32 + lg * 8]);
      af[g].q[1] = *(const uint4*)(&As[cur][m][g * 32 + 16 + lg * 8]);
      #pragma unroll
      for (int t = 0; t < 4; ++t) {
        int j = t * 16 + ln;
        bfr[g][t].q[0] = *(const uint4*)(&Bs[cur][j][g * 32 + lg * 8]);
        bfr[g][t].q[1] = *(const uint4*)(&Bs[cur][j][g * 32 + 16 + lg * 8]);
      }
    }
    #pragma unroll
    for (int g = 0; g < 2; ++g)
      #pragma unroll
      for (int t = 0; t < 4; ++t)
        acc[t] = __builtin_amdgcn_wmma_f32_16x16x32_bf16(
            false, af[g].v, false, bfr[g][t].v, (short)0, acc[t], false, false);

    if (more) {
      *((uint4*)(&As[cur ^ 1][arow][aq * 16]))      = pa0;
      *((uint4*)(&As[cur ^ 1][arow + 64][aq * 16])) = pa1;
      *((uint4*)(&Bs[cur ^ 1][brow][bq * 16]))      = pb;
    }
    __syncthreads();
    cur ^= 1;
  }

  // Epilogue: bias (+ELU), store
  #pragma unroll
  for (int t = 0; t < 4; ++t) {
    int col = colBase + t * 16 + ln;
    float bv = bias[col];
    #pragma unroll
    for (int r = 0; r < 8; ++r) {
      int row = rowBase + wave * 16 + lg * 8 + r;
      float v = acc[t][r] + bv;
      if (ELU) v = (v > 0.0f) ? v : (__expf(v) - 1.0f);
      if (OUT_BF16) ((bf16_t*)Cout)[(size_t)row * N + col] = f2bf(v);
      else          ((float*)Cout)[(size_t)row * N + col]  = v;
    }
  }
}

// ---------------------------------------------------------------------------
// Fused similarity kernel:  S = A @ B^T  (A,B: [N x K] bf16 row-major)
// epilogue: e = exp(2*s); rowsum[i] += sum_j e; colsum[j] += sum_i e.
// MODE 1: full matrix, 2D grid, separate colsum buffer        (S12)
// MODE 2: A==B symmetric, 1D triangular grid (bx<=by tiles);
//         col-partials fold into rowsum; diagonal tiles rows-only (S11,S22)
// Block tile 128x128; 8 waves as 4(M) x 2(N); wave tile 32x64.
// ---------------------------------------------------------------------------
#define SBM 128
#define SBN 128

template <int MODE>
__global__ __launch_bounds__(256) void sim_rowsum_kernel(
    const bf16_t* __restrict__ A, const bf16_t* __restrict__ B,
    float* __restrict__ rowsum, float* __restrict__ colsum,
    int N, int K)
{
  __shared__ __align__(128) bf16_t As[2][SBM][KP];
  __shared__ __align__(128) bf16_t Bs[2][SBN][KP];   // rows of B == cols of B^T

  int bx, by;
  if (MODE == 2) {
    int L = blockIdx.x;                 // triangular decode: bx <= by
    int t = (int)((sqrtf(8.0f * (float)L + 1.0f) - 1.0f) * 0.5f);
    while ((t + 1) * (t + 2) / 2 <= L) ++t;
    while (t * (t + 1) / 2 > L) --t;
    by = t;
    bx = L - t * (t + 1) / 2;
  } else {
    bx = blockIdx.x;
    by = blockIdx.y;
  }
  const bool isDiag = (MODE == 2) && (bx == by);

  const int tid  = threadIdx.x;
  const int wave = tid >> 5;
  const int lane = tid & 31;
  const int lg   = lane >> 4;
  const int ln   = lane & 15;
  const int wm   = wave >> 1;     // 0..3 : 32-row strip
  const int wn   = wave & 1;      // 0..1 : 64-col strip
  const int rowBase = bx * SBM;
  const int colBase = by * SBN;

  // loaders: each tile 128x64 = 512 x 32B slots -> 2 per thread per matrix
  const int rr = tid >> 2, qq = tid & 3;
  const bf16_t* gA0 = A + (size_t)(rowBase + rr) * K + qq * 16;
  const bf16_t* gA1 = gA0 + (size_t)64 * K;
  const bf16_t* gB0 = B + (size_t)(colBase + rr) * K + qq * 16;
  const bf16_t* gB1 = gB0 + (size_t)64 * K;

  v8f acc[2][4];
  #pragma unroll
  for (int mi = 0; mi < 2; ++mi)
    #pragma unroll
    for (int t = 0; t < 4; ++t) acc[mi][t] = (v8f){0,0,0,0,0,0,0,0};

  const int nk = K / KB;
  {
    *((uint4*)(&As[0][rr][qq * 16]))      = *(const uint4*)(gA0);
    *((uint4*)(&As[0][rr + 64][qq * 16])) = *(const uint4*)(gA1);
    *((uint4*)(&Bs[0][rr][qq * 16]))      = *(const uint4*)(gB0);
    *((uint4*)(&Bs[0][rr + 64][qq * 16])) = *(const uint4*)(gB1);
  }
  __syncthreads();

  int cur = 0;
  for (int kk = 0; kk < nk; ++kk) {
    uint4 pa0, pa1, pb0, pb1;
    const bool more = (kk + 1) < nk;
    if (more) {
      pa0 = *(const uint4*)(gA0 + (kk + 1) * KB);
      pa1 = *(const uint4*)(gA1 + (kk + 1) * KB);
      pb0 = *(const uint4*)(gB0 + (kk + 1) * KB);
      pb1 = *(const uint4*)(gB1 + (kk + 1) * KB);
      if (kk + 2 < nk) {
        __builtin_prefetch(gA0 + (kk + 2) * KB, 0, 1);
        __builtin_prefetch(gB0 + (kk + 2) * KB, 0, 1);
      }
    }

    // load BOTH 32-wide fragment groups up front; the 12 LDS loads of group1
    // overlap the 8 WMMAs of group0 (distinct live registers by construction)
    frag16 af[2][2], bfr[2][4];
    #pragma unroll
    for (int g = 0; g < 2; ++g) {
      #pragma unroll
      for (int mi = 0; mi < 2; ++mi) {
        int m = wm * 32 + mi * 16 + ln;
        af[g][mi].q[0] = *(const uint4*)(&As[cur][m][g * 32 + lg * 8]);
        af[g][mi].q[1] = *(const uint4*)(&As[cur][m][g * 32 + 16 + lg * 8]);
      }
      #pragma unroll
      for (int t = 0; t < 4; ++t) {
        int j = wn * 64 + t * 16 + ln;      // lane holds B-row j == S column j
        bfr[g][t].q[0] = *(const uint4*)(&Bs[cur][j][g * 32 + lg * 8]);
        bfr[g][t].q[1] = *(const uint4*)(&Bs[cur][j][g * 32 + 16 + lg * 8]);
      }
    }
    #pragma unroll
    for (int g = 0; g < 2; ++g)
      #pragma unroll
      for (int t = 0; t < 4; ++t)
        #pragma unroll
        for (int mi = 0; mi < 2; ++mi)
          acc[mi][t] = __builtin_amdgcn_wmma_f32_16x16x32_bf16(
              false, af[g][mi].v, false, bfr[g][t].v, (short)0,
              acc[mi][t], false, false);

    if (more) {
      *((uint4*)(&As[cur ^ 1][rr][qq * 16]))      = pa0;
      *((uint4*)(&As[cur ^ 1][rr + 64][qq * 16])) = pa1;
      *((uint4*)(&Bs[cur ^ 1][rr][qq * 16]))      = pb0;
      *((uint4*)(&Bs[cur ^ 1][rr + 64][qq * 16])) = pb1;
    }
    __syncthreads();
    cur ^= 1;
  }

  // exp(2s) + fused row/col reductions (1/TEMP = 2)
  float rowpart[2][8];
  #pragma unroll
  for (int mi = 0; mi < 2; ++mi)
    #pragma unroll
    for (int r = 0; r < 8; ++r) rowpart[mi][r] = 0.0f;

  #pragma unroll
  for (int t = 0; t < 4; ++t) {
    float colpart = 0.0f;
    #pragma unroll
    for (int mi = 0; mi < 2; ++mi) {
      #pragma unroll
      for (int r = 0; r < 8; ++r) {
        float e = __expf(2.0f * acc[mi][t][r]);
        rowpart[mi][r] += e;
        colpart        += e;
      }
    }
    if (MODE == 1 || !isDiag) {           // mirror tile (symm) / true colsum
      colpart += __shfl_xor(colpart, 16, 32);
      if (lg == 0) atomicAdd(&colsum[colBase + wn * 64 + t * 16 + ln], colpart);
    }
  }
  #pragma unroll
  for (int mi = 0; mi < 2; ++mi) {
    #pragma unroll
    for (int r = 0; r < 8; ++r) {
      float v = rowpart[mi][r];
      v += __shfl_xor(v, 1, 32);
      v += __shfl_xor(v, 2, 32);
      v += __shfl_xor(v, 4, 32);
      v += __shfl_xor(v, 8, 32);          // reduce over N within lane group
      if (ln == 0)
        atomicAdd(&rowsum[rowBase + wm * 32 + mi * 16 + lg * 8 + r], v);
    }
  }
}

// ---------------------------------------------------------------------------
// Row normalize (f32 in -> bf16 out), one block (256 thr) per row
// ---------------------------------------------------------------------------
__global__ __launch_bounds__(256) void normalize_rows_kernel(
    const float* __restrict__ Z, bf16_t* __restrict__ Zn, int N)
{
  int row = blockIdx.x;
  const float* zr = Z + (size_t)row * N;
  float s = 0.0f;
  for (int j = threadIdx.x; j < N; j += blockDim.x) { float v = zr[j]; s += v * v; }
  float tot = block_reduce_sum_256(s);
  float inv = 1.0f / fmaxf(sqrtf(tot), 1e-12f);
  for (int j = threadIdx.x; j < N; j += blockDim.x)
    Zn[(size_t)row * N + j] = f2bf(zr[j] * inv);
}

// inverse L2 norms of f32 rows (for raw h1 edge loss)
__global__ __launch_bounds__(256) void row_invnorm_kernel(
    const float* __restrict__ X, float* __restrict__ invn, int N)
{
  int row = blockIdx.x;
  const float* xr = X + (size_t)row * N;
  float s = 0.0f;
  for (int j = threadIdx.x; j < N; j += blockDim.x) { float v = xr[j]; s += v * v; }
  float tot = block_reduce_sum_256(s);
  if (threadIdx.x == 0) invn[row] = 1.0f / fmaxf(sqrtf(tot), 1e-12f);
}

// d[i] = dot(Zn1[i], Zn2[i])  (bf16 inputs, f32 accumulate)
__global__ __launch_bounds__(256) void rowdot_kernel(
    const bf16_t* __restrict__ Z1, const bf16_t* __restrict__ Z2,
    float* __restrict__ d, int N)
{
  int row = blockIdx.x;
  float s = 0.0f;
  for (int j = threadIdx.x; j < N; j += blockDim.x)
    s += bf2f(Z1[(size_t)row * N + j]) * bf2f(Z2[(size_t)row * N + j]);
  float tot = block_reduce_sum_256(s);
  if (threadIdx.x == 0) d[row] = tot;
}

// ---------------------------------------------------------------------------
// Edge kernel: one wave per edge, gathered cosine dot on raw h1.
// POS: store exp(2*sigmoid(cos)); NEG: accumulate sum into negsum.
// ---------------------------------------------------------------------------
template <bool POS>
__global__ __launch_bounds__(256) void edge_kernel(
    const float* __restrict__ h, const float* __restrict__ invn,
    const int* __restrict__ edges, int E, int H,
    float* __restrict__ pos_pexp, float* __restrict__ negsum)
{
  __shared__ float bsum;
  if (!POS) {
    if (threadIdx.x == 0) bsum = 0.0f;
    __syncthreads();
  }
  int wave = threadIdx.x >> 5, lane = threadIdx.x & 31;
  int e = blockIdx.x * 8 + wave;
  float pexp = 0.0f;
  bool valid = (e < E);
  if (valid) {
    int a = edges[e], b = edges[E + e];
    const float4* pa = (const float4*)(h + (size_t)a * H);
    const float4* pb = (const float4*)(h + (size_t)b * H);
    float dot = 0.0f;
    const int nv4 = H >> 2;                     // 128 for H=512
    for (int i = lane; i < nv4; i += 32) {
      float4 x = pa[i], y = pb[i];
      dot += x.x * y.x + x.y * y.y + x.z * y.z + x.w * y.w;
    }
    #pragma unroll
    for (int m = 16; m >= 1; m >>= 1) dot += __shfl_xor(dot, m, 32);
    if (lane == 0) {
      float s   = dot * invn[a] * invn[b];
      float sig = 1.0f / (1.0f + __expf(-s));
      pexp = __expf(2.0f * sig);                // exp(sig / TEMP), TEMP=0.5
      if (POS) pos_pexp[e] = pexp;
    }
  }
  if (!POS) {
    if (lane == 0 && valid) atomicAdd(&bsum, pexp);
    __syncthreads();
    if (threadIdx.x == 0) atomicAdd(negsum, bsum);
  }
}

// ---------------------------------------------------------------------------
// Finalize
// ---------------------------------------------------------------------------
__global__ __launch_bounds__(256) void contrastive_finalize_kernel(
    const float* __restrict__ r11, const float* __restrict__ r12,
    const float* __restrict__ c12, const float* __restrict__ r22,
    const float* __restrict__ d, float* __restrict__ accC, int N)
{
  int i = blockIdx.x * blockDim.x + threadIdx.x;
  float v = 0.0f;
  if (i < N) {
    const float E2 = __expf(2.0f);              // exp(1/TEMP), self-sim term
    float l1 = __logf(r11[i] + r12[i] - E2) - 2.0f * d[i];
    float l2 = __logf(r22[i] + c12[i] - E2) - 2.0f * d[i];
    v = 0.5f * (l1 + l2);
  }
  float tot = block_reduce_sum_256(v);
  if (threadIdx.x == 0) atomicAdd(accC, tot);
}

__global__ __launch_bounds__(256) void edge_finalize_kernel(
    const float* __restrict__ pos_pexp, const float* __restrict__ negsum,
    float* __restrict__ accE, int E)
{
  int i = blockIdx.x * blockDim.x + threadIdx.x;
  float v = 0.0f;
  if (i < E) {
    float p = pos_pexp[i];
    v = __logf(p / (p + negsum[0]) + 1e-15f);
  }
  float tot = block_reduce_sum_256(v);
  if (threadIdx.x == 0) atomicAdd(accE, tot);
}

__global__ void final_kernel(const float* __restrict__ accC,
                             const float* __restrict__ accE,
                             float* __restrict__ out, float invN, float invE)
{
  if (threadIdx.x == 0) out[0] = accC[0] * invN - accE[0] * invE;
}

// ---------------------------------------------------------------------------
// Host launcher
// ---------------------------------------------------------------------------
extern "C" void kernel_launch(void* const* d_in, const int* in_sizes, int n_in,
                              void* d_out, int out_size, void* d_ws, size_t ws_size,
                              hipStream_t stream) {
  const float* h1 = (const float*)d_in[0];
  const float* h2 = (const float*)d_in[1];
  const float* w1 = (const float*)d_in[2];
  const float* b1 = (const float*)d_in[3];
  const float* w2 = (const float*)d_in[4];
  const float* b2 = (const float*)d_in[5];
  const int*   me = (const int*)d_in[6];
  const int*   ne = (const int*)d_in[7];

  const int N = 8192, H = 512, O = 256;
  const int E = in_sizes[6] / 2;

  // workspace carve-up (256B aligned)
  char* base = (char*)d_ws;
  size_t off = 0;
  auto alloc = [&](size_t bytes) -> char* {
    char* p = base + off;
    off = (off + bytes + 255) & ~(size_t)255;
    return p;
  };
  bf16_t* h1b  = (bf16_t*)alloc((size_t)N * H * 2);
  bf16_t* h2b  = (bf16_t*)alloc((size_t)N * H * 2);
  bf16_t* w1t  = (bf16_t*)alloc((size_t)H * O * 2);      // [O x H]
  bf16_t* w2t  = (bf16_t*)alloc((size_t)O * H * 2);      // [H x O]
  bf16_t* T    = (bf16_t*)alloc((size_t)N * O * 2);      // hidden (reused)
  float*  Zf   = (float*) alloc((size_t)N * H * 4);      // proj f32 (reused)
  bf16_t* Zn1  = (bf16_t*)alloc((size_t)N * H * 2);
  bf16_t* Zn2  = (bf16_t*)alloc((size_t)N * H * 2);
  float*  stats = (float*)alloc(((size_t)4 * N + 64) * 4);
  float*  r11 = stats;           float* r12 = stats + N;
  float*  c12 = stats + 2 * N;   float* r22 = stats + 3 * N;
  float*  negsum = stats + 4 * N;
  float*  accC   = stats + 4 * N + 1;
  float*  accE   = stats + 4 * N + 2;
  float*  dvec   = (float*)alloc((size_t)N * 4);
  float*  invn   = (float*)alloc((size_t)N * 4);
  float*  pos_pexp = (float*)alloc((size_t)E * 4);
  (void)ws_size; (void)n_in; (void)out_size;

  // 0) zero accumulators
  zero_kernel<<<(4 * N + 64 + 255) / 256, 256, 0, stream>>>(stats, 4 * N + 64);

  // 1) f32 -> bf16 conversions (+ weight transposes)
  f32_to_bf16_kernel<<<(N * H + 255) / 256, 256, 0, stream>>>(h1, h1b, N * H);
  f32_to_bf16_kernel<<<(N * H + 255) / 256, 256, 0, stream>>>(h2, h2b, N * H);
  transpose_f32_to_bf16_kernel<<<(H * O + 255) / 256, 256, 0, stream>>>(w1, w1t, H, O);
  transpose_f32_to_bf16_kernel<<<(O * H + 255) / 256, 256, 0, stream>>>(w2, w2t, O, H);

  dim3 g1(N / GBM, O / GBN);   // 64 x 4
  dim3 g2(N / GBM, H / GBN);   // 64 x 8

  // 2) projection of h1:  T = elu(h1 @ W1 + b1);  Z = T @ W2 + b2;  Zn1 = norm(Z)
  wmma_gemm_bias_kernel<true,  true ><<<g1, 256, 0, stream>>>(h1b, w1t, b1, T,  N, O, H);
  wmma_gemm_bias_kernel<false, false><<<g2, 256, 0, stream>>>(T,   w2t, b2, Zf, N, H, O);
  normalize_rows_kernel<<<N, 256, 0, stream>>>(Zf, Zn1, H);

  // 3) projection of h2
  wmma_gemm_bias_kernel<true,  true ><<<g1, 256, 0, stream>>>(h2b, w1t, b1, T,  N, O, H);
  wmma_gemm_bias_kernel<false, false><<<g2, 256, 0, stream>>>(T,   w2t, b2, Zf, N, H, O);
  normalize_rows_kernel<<<N, 256, 0, stream>>>(Zf, Zn2, H);

  // 4) diagonal of the "between" similarity
  rowdot_kernel<<<N, 256, 0, stream>>>(Zn1, Zn2, dvec, H);

  // 5) fused similarity row/col sums
  const int NT = N / SBM;                       // 64 tile-rows
  const int triBlocks = NT * (NT + 1) / 2;      // 2080 upper-triangle tiles
  dim3 gs(NT, NT);
  sim_rowsum_kernel<2><<<triBlocks, 256, 0, stream>>>(Zn1, Zn1, r11, r11, N, H);
  sim_rowsum_kernel<1><<<gs,        256, 0, stream>>>(Zn1, Zn2, r12, c12, N, H);
  sim_rowsum_kernel<2><<<triBlocks, 256, 0, stream>>>(Zn2, Zn2, r22, r22, N, H);

  // 6) edge reconstruction loss pieces on raw h1
  row_invnorm_kernel<<<N, 256, 0, stream>>>(h1, invn, H);
  edge_kernel<true ><<<(E + 7) / 8, 256, 0, stream>>>(h1, invn, me, E, H, pos_pexp, nullptr);
  edge_kernel<false><<<(E + 7) / 8, 256, 0, stream>>>(h1, invn, ne, E, H, nullptr, negsum);

  // 7) finalize
  contrastive_finalize_kernel<<<(N + 255) / 256, 256, 0, stream>>>(r11, r12, c12, r22,
                                                                   dvec, accC, N);
  edge_finalize_kernel<<<(E + 255) / 256, 256, 0, stream>>>(pos_pexp, negsum, accE, E);
  final_kernel<<<1, 32, 0, stream>>>(accC, accE, (float*)d_out, 1.0f / N, 1.0f / E);
}
